// ExpertChoiceTopKRouter_23656679867557
// MI455X (gfx1250) — matrix operand order repaired and checked by reading, not compiled, not run on hardware
//
#include <hip/hip_runtime.h>
#include <hip/hip_bf16.h>
#include <stdint.h>

// ---------------------------------------------------------------------------
// ExpertChoiceTopKRouter for MI455X (gfx1250)
//   x: [T=16384, D=2048] f32,  W: [E=8, D=2048] f32
//   scores = sigmoid(x @ W^T) -> [E, T]; per-expert top-2048 (desc, idx ties)
//
// HBM-bound: x = 134 MB read once @ 23.3 TB/s ~ 6 us floor. Kernel 1 streams
// x once through v_wmma_f32_16x16x32_bf16 (E padded to N=16), with conversion
// cost cut to 1 v_perm_b32 per float pair so VALU can't cap the HBM stream.
// Kernel 2 does exact top-k: 4-pass radix-select + 4096-key bitonic in LDS.
// ---------------------------------------------------------------------------

typedef __attribute__((ext_vector_type(16))) __bf16 v16bf;
typedef __attribute__((ext_vector_type(8)))  float  v8f;

#define TOKENS   16384
#define DIM      2048
#define NEXP     8
#define NEXP_PAD 16     // WMMA N dimension
#define KSEL     2048   // tokens_per_expert
#define CAND_CAP 4096   // candidate pool for final sort (pow2 >= KSEL + ties)

// Pack two f32 into one dword of 2x bf16 (truncation) -- single v_perm_b32.
static __device__ __forceinline__ unsigned pk2bf(float f0, float f1) {
#if __has_builtin(__builtin_amdgcn_perm)
    return __builtin_amdgcn_perm(__float_as_uint(f1), __float_as_uint(f0),
                                 0x07060302u);
#else
    return (__float_as_uint(f1) & 0xFFFF0000u) | (__float_as_uint(f0) >> 16);
#endif
}

// f32 -> bf16 (RNE) as raw u16 -- used once for the tiny W pre-conversion.
static __device__ __forceinline__ unsigned short f2bf_rne(float f) {
    unsigned u = __float_as_uint(f);
    return (unsigned short)((u + 0x7FFFu + ((u >> 16) & 1u)) >> 16);
}

// ---------------------------------------------------------------------------
// Kernel 0: W [8,2048] f32 -> Wbf [16,2048] bf16, experts 8..15 zero-padded.
// ---------------------------------------------------------------------------
__global__ void __launch_bounds__(256)
router_wconv(const float* __restrict__ W, unsigned short* __restrict__ Wbf) {
    int idx = blockIdx.x * blockDim.x + threadIdx.x;   // [0, 16*2048)
    if (idx >= NEXP_PAD * DIM) return;
    int row = idx / DIM;
    float v = (row < NEXP) ? W[(size_t)row * DIM + (idx - row * DIM)] : 0.0f;
    Wbf[idx] = f2bf_rne(v);
}

// ---------------------------------------------------------------------------
// Kernel 1: logits via WMMA + fp32 sigmoid -> scores[E][T] in workspace.
// One wave owns a 16-token x 16-expert tile; K-loop over 2048 in steps of 32.
// ---------------------------------------------------------------------------
__global__ void __launch_bounds__(256)
router_gemm_sigmoid(const float* __restrict__ x,
                    const unsigned short* __restrict__ Wbf,
                    float* __restrict__ scores) {
    const int lane = threadIdx.x & 31;
    const int wave = threadIdx.x >> 5;
    const int tile = blockIdx.x * (256 >> 5) + wave;   // 16-token tile id
    const int row0 = tile * 16;
    const int half = lane >> 4;          // 0: lanes 0-15, 1: lanes 16-31
    const int l16  = lane & 15;

    const float* aBase = x + (size_t)(row0 + l16) * DIM + half * 8;
    const unsigned short* bBase = Wbf + (size_t)l16 * DIM + half * 8;

    v8f c = {};                          // 16x16 f32 accumulator

    for (int k0 = 0; k0 < DIM; k0 += 32) {
        // ---- A fragment: 16x32 bf16 of x (ISA A-layout) ----
        // lane half h: K = k0 + h*8 + [0..7]  and  k0 + 16 + h*8 + [0..7]
        float fa[16];
        *reinterpret_cast<float4*>(&fa[0])  = *reinterpret_cast<const float4*>(aBase + k0);
        *reinterpret_cast<float4*>(&fa[4])  = *reinterpret_cast<const float4*>(aBase + k0 + 4);
        *reinterpret_cast<float4*>(&fa[8])  = *reinterpret_cast<const float4*>(aBase + k0 + 16);
        *reinterpret_cast<float4*>(&fa[12]) = *reinterpret_cast<const float4*>(aBase + k0 + 20);

        union { unsigned u[8]; v16bf v; } au;
#pragma unroll
        for (int i = 0; i < 8; ++i) au.u[i] = pk2bf(fa[2 * i], fa[2 * i + 1]);

        // ---- B fragment: 32x16 bf16 = Wbf^T, column (expert) per lane ----
        // lane half h: K = k0 + h*8 + [0..7]  and  k0 + 16 + h*8 + [0..7]
        union { uint4 q[2]; v16bf v; } bu;
        bu.q[0] = *reinterpret_cast<const uint4*>(bBase + k0);
        bu.q[1] = *reinterpret_cast<const uint4*>(bBase + k0 + 16);

        c = __builtin_amdgcn_wmma_f32_16x16x32_bf16(
                /*neg_a=*/false, au.v, /*neg_b=*/false, bu.v,
                /*c_mod=*/(short)0, c, /*reuse_a=*/false, /*reuse_b=*/false);
    }

    // C layout: VGPR r holds M = r (+8 for lanes 16-31), N = lane&15.
    // => lane (expert e=l16) owns 8 consecutive tokens: row0 + half*8 + r
    if (l16 < NEXP) {
        const int tok0 = row0 + half * 8;
        float out8[8];
#pragma unroll
        for (int r = 0; r < 8; ++r)
            out8[r] = 1.0f / (1.0f + __expf(-c[r]));   // fp32 sigmoid
        float* dst = scores + (size_t)l16 * TOKENS + tok0;
        *reinterpret_cast<float4*>(dst)     = *reinterpret_cast<float4*>(&out8[0]);
        *reinterpret_cast<float4*>(dst + 4) = *reinterpret_cast<float4*>(&out8[4]);
    }
}

// ---------------------------------------------------------------------------
// Kernel 2: exact per-expert top-k.
//   1) score bit-patterns (positive floats -> order-preserving u32) into LDS
//   2) 4-pass MSB-first radix select (256-bin LDS histogram) -> exact k-th
//      threshold bits
//   3) compact all elements >= thr as u64 keys (score<<32 | ~idx)
//   4) bitonic sort CAND_CAP=4096 keys desc; ~idx => lowest index wins ties
//   5) emit first KSEL scores + indices
// LDS: 64KB bits + 32KB candidates + 1KB hist + ctl  (< 320KB/WGP on CDNA5)
// ---------------------------------------------------------------------------
__global__ void __launch_bounds__(1024)
router_topk(const float* __restrict__ scores,
            float* __restrict__ outScores,
            int* __restrict__ outIdx) {
    extern __shared__ unsigned char smem[];
    unsigned*            sbits = (unsigned*)smem;                       // 16384
    unsigned long long*  cand  = (unsigned long long*)(smem + 65536);   // 4096
    unsigned*            hist  = (unsigned*)(smem + 65536 + 32768);     // 256
    unsigned*            ctl   = hist + 256;                            // 4

    const int e   = blockIdx.x;
    const int tid = threadIdx.x;
    const int nth = blockDim.x;

    // 1) load score bits
    const float* s = scores + (size_t)e * TOKENS;
    for (int i = tid; i < TOKENS; i += nth)
        sbits[i] = __float_as_uint(s[i]);
    // init candidate pool while we're at it
    for (int i = tid; i < CAND_CAP; i += nth)
        cand[i] = 0ull;
    if (tid == 0) ctl[3] = 0u;
    __syncthreads();

    // 2) radix select: find threshold bits of the KSEL-th largest
    unsigned prefix    = 0;
    unsigned remaining = KSEL;
    for (int pass = 0; pass < 4; ++pass) {
        const int shift = 24 - 8 * pass;
        for (int i = tid; i < 256; i += nth) hist[i] = 0u;
        __syncthreads();
        for (int i = tid; i < TOKENS; i += nth) {
            unsigned v  = sbits[i];
            unsigned hi = (pass == 0) ? 0u : (v >> (shift + 8));
            if (hi == prefix)
                atomicAdd(&hist[(v >> shift) & 0xFFu], 1u);
        }
        __syncthreads();
        if (tid == 0) {
            unsigned acc = 0, sel = 0;
            for (int b = 255; b >= 0; --b) {
                unsigned cb = hist[b];
                if (acc + cb >= remaining) { sel = (unsigned)b; break; }
                acc += cb;
            }
            ctl[0] = acc;   // count strictly above selected bin
            ctl[1] = sel;   // selected digit
        }
        __syncthreads();
        remaining -= ctl[0];
        prefix     = (prefix << 8) | ctl[1];
        __syncthreads();
    }
    const unsigned thr = prefix;   // exact bits of the k-th largest score

    // 3) compact candidates (>= thr); order fixed by the sort below
    for (int i = tid; i < TOKENS; i += nth) {
        unsigned v = sbits[i];
        if (v >= thr) {
            unsigned pos = atomicAdd(&ctl[3], 1u);
            if (pos < CAND_CAP)
                cand[pos] = ((unsigned long long)v << 32) |
                            (unsigned long long)(0xFFFFFFFFu - (unsigned)i);
        }
    }
    __syncthreads();

    // 4) bitonic sort CAND_CAP keys, descending (zero-pads sink)
    for (unsigned k = 2; k <= (unsigned)CAND_CAP; k <<= 1) {
        for (unsigned j = k >> 1; j > 0; j >>= 1) {
            for (unsigned i = tid; i < (unsigned)CAND_CAP; i += nth) {
                unsigned ixj = i ^ j;
                if (ixj > i) {
                    unsigned long long a = cand[i];
                    unsigned long long b = cand[ixj];
                    bool descBlock = ((i & k) == 0);
                    if (descBlock ? (a < b) : (a > b)) {
                        cand[i]   = b;
                        cand[ixj] = a;
                    }
                }
            }
            __syncthreads();
        }
    }

    // 5) emit top KSEL
    for (int i = tid; i < KSEL; i += nth) {
        unsigned long long kk = cand[i];
        outScores[(size_t)e * KSEL + i] = __uint_as_float((unsigned)(kk >> 32));
        outIdx[(size_t)e * KSEL + i]    = (int)(0xFFFFFFFFu - (unsigned)kk);
    }
}

// ---------------------------------------------------------------------------
extern "C" void kernel_launch(void* const* d_in, const int* in_sizes, int n_in,
                              void* d_out, int out_size, void* d_ws, size_t ws_size,
                              hipStream_t stream) {
    const float* x = (const float*)d_in[0];   // [T, D] f32
    const float* W = (const float*)d_in[1];   // [E, D] f32

    float*          scores = (float*)d_ws;                         // 512 KB
    unsigned short* Wbf    = (unsigned short*)((char*)d_ws + (size_t)NEXP * TOKENS * 4);
                                                                   // + 64 KB
    float* outScores = (float*)d_out;                              // [E, KSEL]
    int*   outIdx    = (int*)((float*)d_out + NEXP * KSEL);        // [E, KSEL]

    // Kernel 0: pad+convert W to bf16 (16 x 2048)
    router_wconv<<<(NEXP_PAD * DIM + 255) / 256, 256, 0, stream>>>(W, Wbf);

    // Kernel 1: 1024 tiles of 16 tokens, 8 waves/block -> 128 blocks
    const int wavesPerBlock = 256 / 32;
    const int numTiles = TOKENS / 16;
    router_gemm_sigmoid<<<numTiles / wavesPerBlock, 256, 0, stream>>>(x, Wbf, scores);

    // Kernel 2: one block per expert; dynamic LDS = 64K + 32K + 1K + ctl
    const size_t lds2 = 65536 + 32768 + 1024 + 64;
    router_topk<<<NEXP, 1024, lds2, stream>>>(scores, outScores, outIdx);
}
